// AutoregressivePrior_70875550318618
// MI455X (gfx1250) — matrix operand order repaired and checked by reading, not compiled, not run on hardware
//
#include <hip/hip_runtime.h>
#include <hip/hip_bf16.h>

// -------- CDNA5 (gfx1250) wave32 WMMA GEMM pipeline for autoregressive LSTM prior -----
// f16 WMMA (v_wmma_f32_16x16x32_f16), fp32 accumulation, fused LSTM / head epilogues.
// K-tiles staged with GLOBAL_LOAD_ASYNC_TO_LDS_B128 into a ping-pong LDS buffer,
// synchronized with s_wait_asynccnt (ASYNCcnt) + workgroup barriers.

typedef __attribute__((ext_vector_type(16))) _Float16 v16h;
typedef __attribute__((ext_vector_type(8)))  _Float16 v8h;
typedef __attribute__((ext_vector_type(8)))  float    v8f;

#define B_TOK   16384
#define SCENE   256
#define FEAT    256
#define HID     1024
#define NSLOT   8
#define LDS_STR 40          // padded LDS row stride (halves) -> conflict-free b128 reads

struct StageLDS { _Float16 A[128 * LDS_STR]; _Float16 Bm[128 * LDS_STR]; };
struct SmemU { union { StageLDS st[2]; float C[64 * 132]; } u; };

__device__ __forceinline__ float sigm_f(float x) { return 1.0f / (1.0f + expf(-x)); }
__device__ __forceinline__ float softplus_f(float x) { return (x > 20.0f) ? x : log1pf(expf(x)); }
__device__ __forceinline__ float gelu_f(float x) {
  return 0.5f * x * (1.0f + erff(x * 0.70710678118654752440f));
}

// Async-copy 32 contiguous bytes (global -> LDS), no VGPR staging. Same INST_OFFSET is
// applied to both the global address and the LDS destination per the ISA.
__device__ __forceinline__ void async_cp32(unsigned lds_addr, const void* gaddr) {
  asm volatile("global_load_async_to_lds_b128 %0, %1, off\n\t"
               "global_load_async_to_lds_b128 %0, %1, off offset:16"
               :: "v"(lds_addr), "v"(gaddr) : "memory");
}
__device__ __forceinline__ unsigned lds_off(const void* p) {
  return (unsigned)(unsigned long long)p;   // low 32 bits of generic ptr = LDS offset
}

// A fragment (16x32 f16, MxK): lanes 0-15 row M=l, K {0..7,16..23}; lanes 16-31 K {8..15,24..31}
__device__ __forceinline__ v16h frag_a(const _Float16* sA, int lane, int mbase) {
  union { v16h v; v8h h[2]; } f;
  const int r  = mbase + (lane & 15);
  const int k0 = (lane >> 4) * 8;
  const _Float16* p = sA + r * LDS_STR + k0;
  f.h[0] = *(const v8h*)(p);
  f.h[1] = *(const v8h*)(p + 16);
  return f.v;
}

// B fragment (32x16 f16, KxN): lanes 0-15 col N=l, K {0..7,16..23}; lanes 16-31 K {8..15,24..31}
__device__ __forceinline__ v16h frag_b(const _Float16* sB, int lane, int nbase) {
  union { v16h v; v8h h[2]; } f;
  const int c  = nbase + (lane & 15);
  const int k0 = (lane >> 4) * 8;
  const _Float16* p = sB + c * LDS_STR + k0;
  f.h[0] = *(const v8h*)(p);
  f.h[1] = *(const v8h*)(p + 16);
  return f.v;
}

// one K=32 step: 8 waves tile 128x128 as 4(M)x2(N) waves, each wave 2x4 wmma tiles
__device__ __forceinline__ void mma_step(const StageLDS& st, int tid, v8f (&acc)[2][4]) {
  const int lane = tid & 31;
  const int wm = ((tid >> 5) & 3) * 32;
  const int wn = (tid >> 7) * 64;
  v16h a0 = frag_a(st.A, lane, wm);
  v16h a1 = frag_a(st.A, lane, wm + 16);
  v16h b0 = frag_b(st.Bm, lane, wn);
  v16h b1 = frag_b(st.Bm, lane, wn + 16);
  v16h b2 = frag_b(st.Bm, lane, wn + 32);
  v16h b3 = frag_b(st.Bm, lane, wn + 48);
  acc[0][0] = __builtin_amdgcn_wmma_f32_16x16x32_f16(false, a0, false, b0, (short)0, acc[0][0], false, false);
  acc[0][1] = __builtin_amdgcn_wmma_f32_16x16x32_f16(false, a0, false, b1, (short)0, acc[0][1], false, false);
  acc[0][2] = __builtin_amdgcn_wmma_f32_16x16x32_f16(false, a0, false, b2, (short)0, acc[0][2], false, false);
  acc[0][3] = __builtin_amdgcn_wmma_f32_16x16x32_f16(false, a0, false, b3, (short)0, acc[0][3], false, false);
  acc[1][0] = __builtin_amdgcn_wmma_f32_16x16x32_f16(false, a1, false, b0, (short)0, acc[1][0], false, false);
  acc[1][1] = __builtin_amdgcn_wmma_f32_16x16x32_f16(false, a1, false, b1, (short)0, acc[1][1], false, false);
  acc[1][2] = __builtin_amdgcn_wmma_f32_16x16x32_f16(false, a1, false, b2, (short)0, acc[1][2], false, false);
  acc[1][3] = __builtin_amdgcn_wmma_f32_16x16x32_f16(false, a1, false, b3, (short)0, acc[1][3], false, false);
}

// spill one 64-row half of the 128x128 accumulators into the LDS exchange buffer.
// C/D layout: lanes 0-15 hold N=l, M=v; lanes 16-31 hold N=l-16, M=v+8.
__device__ __forceinline__ void store_half(SmemU& sm, int tid, const v8f (&acc)[2][4], int half) {
  const int lane = tid & 31;
  const int wave = tid >> 5;
  const int wmW = wave & 3;
  const int wn = (wave >> 2) * 64;
  if ((wmW >> 1) != half) return;
  const int nl = lane & 15;
  const int mofs = (lane >> 4) * 8;
#pragma unroll
  for (int i = 0; i < 2; ++i) {
    const int mrow = (wmW & 1) * 32 + i * 16 + mofs;
#pragma unroll
    for (int j = 0; j < 4; ++j) {
      float* cp = &sm.u.C[mrow * 132 + wn + j * 16 + nl];
      v8f a = acc[i][j];
#pragma unroll
      for (int v = 0; v < 8; ++v) cp[v * 132] = a[v];
    }
  }
}

// ---------------------------------------------------------------- utility kernels
__global__ __launch_bounds__(256) void f32_to_f16_kernel(const float* __restrict__ src,
                                                         _Float16* __restrict__ dst, int n) {
  int i = blockIdx.x * 256 + threadIdx.x;
  if (i < n) dst[i] = (_Float16)src[i];
}

__global__ __launch_bounds__(256) void zero_state_kernel(_Float16* __restrict__ h,
                                                         float* __restrict__ c, int n) {
  int i = blockIdx.x * 256 + threadIdx.x;
  if (i < n) { h[i] = (_Float16)0.0f; c[i] = 0.0f; }
}

// ---------------------------------------------------------------- encoder: GELU(S We^T + be)
__global__ __launch_bounds__(256) void encoder_kernel(const _Float16* __restrict__ S16,
                                                      const _Float16* __restrict__ We16,
                                                      const float* __restrict__ be,
                                                      _Float16* __restrict__ X16) {
  __shared__ SmemU sm;
  const int tid = threadIdx.x;
  const int m0 = blockIdx.x * 128;
  const int n0 = blockIdx.y * 128;
  const int lrow = tid >> 1;
  const int lk = (tid & 1) * 16;

  const v8f vzero = {0.f,0.f,0.f,0.f,0.f,0.f,0.f,0.f};
  v8f acc[2][4];
#pragma unroll
  for (int i = 0; i < 2; ++i)
#pragma unroll
    for (int j = 0; j < 4; ++j) acc[i][j] = vzero;

  auto issue = [&](int kt) {
    const int kb = kt * 32;
    const _Float16* pa = S16 + (size_t)(m0 + lrow) * SCENE + kb + lk;
    const _Float16* pb = We16 + (size_t)(n0 + lrow) * SCENE + kb + lk;
    StageLDS* buf = &sm.u.st[kt & 1];
    async_cp32(lds_off(buf->A + lrow * LDS_STR + lk), pa);
    async_cp32(lds_off(buf->Bm + lrow * LDS_STR + lk), pb);
  };
  issue(0);
  const int KT = SCENE / 32;
  for (int kt = 0; kt < KT; ++kt) {
    if (kt + 1 < KT) { issue(kt + 1); asm volatile("s_wait_asynccnt 0x4" ::: "memory"); }
    else             { asm volatile("s_wait_asynccnt 0x0" ::: "memory"); }
    __syncthreads();
    mma_step(sm.u.st[kt & 1], tid, acc);
    __syncthreads();
  }
  for (int half = 0; half < 2; ++half) {
    store_half(sm, tid, acc, half);
    __syncthreads();
#pragma unroll
    for (int it = 0; it < 32; ++it) {
      const int idx = tid + it * 256;
      const int m = idx >> 7;
      const int n = idx & 127;
      const float v = sm.u.C[m * 132 + n] + be[n0 + n];
      X16[(size_t)(m0 + half * 64 + m) * FEAT + n0 + n] = (_Float16)gelu_f(v);
    }
    __syncthreads();
  }
}

// ------------------------------------------- gates GEMM + fused LSTM cell (i,f,g,o quadrants)
__global__ __launch_bounds__(256) void gates_kernel(const _Float16* __restrict__ X16,
                                                    const _Float16* __restrict__ Hin,
                                                    const _Float16* __restrict__ Wih16,
                                                    const _Float16* __restrict__ Whh16,
                                                    const float* __restrict__ bih,
                                                    const float* __restrict__ bhh,
                                                    float* __restrict__ Cf,
                                                    _Float16* __restrict__ Hout) {
  __shared__ SmemU sm;
  const int tid = threadIdx.x;
  const int m0 = blockIdx.x * 128;
  const int n0 = blockIdx.y * 32;          // column base within each gate quadrant
  const int lrow = tid >> 1;
  const int lk = (tid & 1) * 16;
  // tile-local B column c -> gate row (c>>5)*HID + n0 + (c&31)
  const int wr = (lrow >> 5) * HID + n0 + (lrow & 31);

  const v8f vzero = {0.f,0.f,0.f,0.f,0.f,0.f,0.f,0.f};
  v8f acc[2][4];
#pragma unroll
  for (int i = 0; i < 2; ++i)
#pragma unroll
    for (int j = 0; j < 4; ++j) acc[i][j] = vzero;

  auto issue = [&](int kt) {
    const _Float16* A; const _Float16* W; int lda; int kb;
    if (kt < 8) { A = X16; W = Wih16; lda = FEAT; kb = kt * 32; }
    else        { A = Hin; W = Whh16; lda = HID;  kb = (kt - 8) * 32; }
    const _Float16* pa = A + (size_t)(m0 + lrow) * lda + kb + lk;
    const _Float16* pb = W + (size_t)wr * lda + kb + lk;
    StageLDS* buf = &sm.u.st[kt & 1];
    async_cp32(lds_off(buf->A + lrow * LDS_STR + lk), pa);
    async_cp32(lds_off(buf->Bm + lrow * LDS_STR + lk), pb);
  };
  issue(0);
  const int KT = 8 + HID / 32;             // x·Wih^T then h·Whh^T into same accumulators
  for (int kt = 0; kt < KT; ++kt) {
    if (kt + 1 < KT) { issue(kt + 1); asm volatile("s_wait_asynccnt 0x4" ::: "memory"); }
    else             { asm volatile("s_wait_asynccnt 0x0" ::: "memory"); }
    __syncthreads();
    mma_step(sm.u.st[kt & 1], tid, acc);
    __syncthreads();
  }
  for (int half = 0; half < 2; ++half) {
    store_half(sm, tid, acc, half);
    __syncthreads();
#pragma unroll
    for (int it = 0; it < 8; ++it) {
      const int idx = tid + it * 256;
      const int m = idx >> 5;
      const int j = idx & 31;
      const float gi = sm.u.C[m * 132 + j]      + bih[n0 + j]           + bhh[n0 + j];
      const float gf = sm.u.C[m * 132 + 32 + j] + bih[HID + n0 + j]     + bhh[HID + n0 + j];
      const float gg = sm.u.C[m * 132 + 64 + j] + bih[2 * HID + n0 + j] + bhh[2 * HID + n0 + j];
      const float go = sm.u.C[m * 132 + 96 + j] + bih[3 * HID + n0 + j] + bhh[3 * HID + n0 + j];
      const size_t gidx = (size_t)(m0 + half * 64 + m) * HID + n0 + j;
      const float cold = Cf[gidx];
      const float cn = sigm_f(gf) * cold + sigm_f(gi) * tanhf(gg);
      Cf[gidx] = cn;
      Hout[gidx] = (_Float16)(sigm_f(go) * tanhf(cn));
    }
    __syncthreads();
  }
}

// --------------------------------- heads GEMM + fused softplus / rsample epilogue
__global__ __launch_bounds__(256) void heads_kernel(const _Float16* __restrict__ Hin,
                                                    const _Float16* __restrict__ Wmu16,
                                                    const _Float16* __restrict__ Wsp16,
                                                    const float* __restrict__ bmu,
                                                    const float* __restrict__ bsp,
                                                    const float* __restrict__ eps_t,
                                                    float* __restrict__ out_z,
                                                    float* __restrict__ out_mu,
                                                    float* __restrict__ out_sg,
                                                    _Float16* __restrict__ X16) {
  __shared__ SmemU sm;
  const int tid = threadIdx.x;
  const int m0 = blockIdx.x * 128;
  const int n0 = blockIdx.y * 64;          // feature base (mu cols 0..63, sp cols 64..127 of tile)
  const int lrow = tid >> 1;
  const int lk = (tid & 1) * 16;
  const _Float16* Wrow = (lrow < 64) ? (Wmu16 + (size_t)(n0 + lrow) * HID)
                                     : (Wsp16 + (size_t)(n0 + lrow - 64) * HID);

  const v8f vzero = {0.f,0.f,0.f,0.f,0.f,0.f,0.f,0.f};
  v8f acc[2][4];
#pragma unroll
  for (int i = 0; i < 2; ++i)
#pragma unroll
    for (int j = 0; j < 4; ++j) acc[i][j] = vzero;

  auto issue = [&](int kt) {
    const int kb = kt * 32;
    const _Float16* pa = Hin + (size_t)(m0 + lrow) * HID + kb + lk;
    const _Float16* pb = Wrow + kb + lk;
    StageLDS* buf = &sm.u.st[kt & 1];
    async_cp32(lds_off(buf->A + lrow * LDS_STR + lk), pa);
    async_cp32(lds_off(buf->Bm + lrow * LDS_STR + lk), pb);
  };
  issue(0);
  const int KT = HID / 32;
  for (int kt = 0; kt < KT; ++kt) {
    if (kt + 1 < KT) { issue(kt + 1); asm volatile("s_wait_asynccnt 0x4" ::: "memory"); }
    else             { asm volatile("s_wait_asynccnt 0x0" ::: "memory"); }
    __syncthreads();
    mma_step(sm.u.st[kt & 1], tid, acc);
    __syncthreads();
  }
  for (int half = 0; half < 2; ++half) {
    store_half(sm, tid, acc, half);
    __syncthreads();
#pragma unroll
    for (int it = 0; it < 16; ++it) {
      const int idx = tid + it * 256;
      const int m = idx >> 6;
      const int j = idx & 63;
      const float mu = sm.u.C[m * 132 + j]      + bmu[n0 + j];
      const float sp = sm.u.C[m * 132 + 64 + j] + bsp[n0 + j];
      const float sg = softplus_f(sp);         // T_TEMP = 1
      const int mg = m0 + half * 64 + m;
      const size_t oidx = (size_t)mg * FEAT + n0 + j;
      const float z = mu + sg * eps_t[oidx];
      out_z[oidx]  = z;
      out_mu[oidx] = mu;
      out_sg[oidx] = sg;
      X16[oidx] = (_Float16)z;                 // next-step LSTM input
    }
    __syncthreads();
  }
}

// ---------------------------------------------------------------- host driver
extern "C" void kernel_launch(void* const* d_in, const int* in_sizes, int n_in,
                              void* d_out, int out_size, void* d_ws, size_t ws_size,
                              hipStream_t stream) {
  (void)in_sizes; (void)n_in; (void)out_size; (void)ws_size;
  const float* S   = (const float*)d_in[0];
  const float* eps = (const float*)d_in[1];
  const float* We  = (const float*)d_in[3];
  const float* be  = (const float*)d_in[4];
  const float* Wih = (const float*)d_in[5];
  const float* Whh = (const float*)d_in[6];
  const float* bih = (const float*)d_in[7];
  const float* bhh = (const float*)d_in[8];
  const float* Wmu = (const float*)d_in[9];
  const float* bmu = (const float*)d_in[10];
  const float* Wsp = (const float*)d_in[11];
  const float* bsp = (const float*)d_in[12];
  float* out = (float*)d_out;

  char* ws = (char*)d_ws;
  size_t off = 0;
  auto take = [&](size_t bytes) -> char* {
    char* p = ws + off;
    off += (bytes + 255) & ~(size_t)255;
    return p;
  };
  _Float16* We16  = (_Float16*)take((size_t)FEAT * SCENE * 2);
  _Float16* Wih16 = (_Float16*)take((size_t)4 * HID * FEAT * 2);
  _Float16* Whh16 = (_Float16*)take((size_t)4 * HID * HID * 2);
  _Float16* Wmu16 = (_Float16*)take((size_t)FEAT * HID * 2);
  _Float16* Wsp16 = (_Float16*)take((size_t)FEAT * HID * 2);
  _Float16* S16   = (_Float16*)take((size_t)B_TOK * SCENE * 2);
  _Float16* X16   = (_Float16*)take((size_t)B_TOK * FEAT * 2);
  _Float16* Ha    = (_Float16*)take((size_t)B_TOK * HID * 2);
  _Float16* Hb    = (_Float16*)take((size_t)B_TOK * HID * 2);
  float*    Cf    = (float*)   take((size_t)B_TOK * HID * 4);

  auto cvt = [&](const float* s, _Float16* d, int n) {
    f32_to_f16_kernel<<<(n + 255) / 256, 256, 0, stream>>>(s, d, n);
  };
  cvt(We,  We16,  FEAT * SCENE);
  cvt(Wih, Wih16, 4 * HID * FEAT);
  cvt(Whh, Whh16, 4 * HID * HID);
  cvt(Wmu, Wmu16, FEAT * HID);
  cvt(Wsp, Wsp16, FEAT * HID);
  cvt(S,   S16,   B_TOK * SCENE);
  {
    const int n = B_TOK * HID;
    zero_state_kernel<<<(n + 255) / 256, 256, 0, stream>>>(Ha, Cf, n);
  }

  encoder_kernel<<<dim3(B_TOK / 128, FEAT / 128), 256, 0, stream>>>(S16, We16, be, X16);

  const size_t slab = (size_t)B_TOK * FEAT;
  for (int s = 0; s < NSLOT; ++s) {
    const _Float16* hin = (s & 1) ? Hb : Ha;
    _Float16* hout      = (s & 1) ? Ha : Hb;
    gates_kernel<<<dim3(B_TOK / 128, HID / 32), 256, 0, stream>>>(
        X16, hin, Wih16, Whh16, bih, bhh, Cf, hout);
    heads_kernel<<<dim3(B_TOK / 128, FEAT / 64), 256, 0, stream>>>(
        hout, Wmu16, Wsp16, bmu, bsp,
        eps + (size_t)s * slab,
        out + (size_t)s * slab,
        out + (size_t)(NSLOT + s) * slab,
        out + (size_t)(2 * NSLOT + s) * slab,
        X16);
  }
}
